// WideEPMoEComputeSimulator_35218731827682
// MI455X (gfx1250) — matrix operand order repaired and checked by reading, not compiled, not run on hardware
//
#include <hip/hip_runtime.h>

// ---------------- problem constants (from reference setup_inputs) ------------
static constexpr int T_ = 16384;   // tokens
static constexpr int H_ = 1024;    // hidden
static constexpr int I_ = 512;     // intermediate
static constexpr int E_ = 64;      // experts
static constexpr int C_ = 1024;    // capacity
// top_k == 2 (d_in[4]), hardcoded per reference setup.

typedef __attribute__((ext_vector_type(16))) __bf16 v16bf;
typedef __attribute__((ext_vector_type(8)))  float  v8f;
typedef __attribute__((ext_vector_type(4)))  int    v4i;

// ---- CDNA5 async global->LDS path (guarded; falls back to sync copies) ------
#if defined(__HIP_DEVICE_COMPILE__) && \
    __has_builtin(__builtin_amdgcn_global_load_async_to_lds_b128) && \
    __has_builtin(__builtin_amdgcn_s_wait_asynccnt)
#define USE_ASYNC_LDS 1
#else
#define USE_ASYNC_LDS 0
#endif

// copy 16B global -> LDS (async: tracked by ASYNCcnt, no VGPR round trip)
// builtin signature (from diagnostics): (v4i AS1*, v4i AS3*, imm int, imm int)
__device__ __forceinline__ void stage16(const unsigned short* g, unsigned short* l) {
#if USE_ASYNC_LDS
    __builtin_amdgcn_global_load_async_to_lds_b128(
        (__attribute__((address_space(1))) v4i*)g,
        (__attribute__((address_space(3))) v4i*)l, 0, 0);
#else
    *(uint4*)l = *(const uint4*)g;
#endif
}
__device__ __forceinline__ void wait_async0() {
#if USE_ASYNC_LDS
    __builtin_amdgcn_s_wait_asynccnt(0);   // s_wait_asynccnt 0
#endif
}

// f32 -> bf16 (round-to-nearest-even), pure integer
__device__ __forceinline__ unsigned short f32_to_bf16(float f) {
    unsigned u = __float_as_uint(f);
    u += 0x7FFFu + ((u >> 16) & 1u);
    return (unsigned short)(u >> 16);
}

// ---- WMMA fragment builders -------------------------------------------------
// LDS tiles K-contiguous, row stride 48 ushort (96B) so every 16B chunk is
// aligned. A 16x32 bf16 (ISA 7.12.2): lane holds row M=L&15, khalf=L>>4 selects
// K runs {khalf*8..+7} and {16+khalf*8..+7}. B 32x16: col N=L&15, same K
// runs {khalf*8..+7} and {16+khalf*8..+7}.
__device__ __forceinline__ v16bf load_a_frag(const unsigned short* lds, int row, int khalf) {
    const unsigned short* p = lds + row * 48;
    union { uint4 q[2]; v16bf v; } u;
    u.q[0] = *(const uint4*)(p + khalf * 8);
    u.q[1] = *(const uint4*)(p + 16 + khalf * 8);
    return u.v;
}
__device__ __forceinline__ v16bf load_b_frag(const unsigned short* lds, int col, int khalf) {
    const unsigned short* p = lds + col * 48;
    union { uint4 q[2]; v16bf v; } u;
    u.q[0] = *(const uint4*)(p + khalf * 8);
    u.q[1] = *(const uint4*)(p + 16 + khalf * 8);
    return u.v;
}
__device__ __forceinline__ v8f wmma_bf16(v16bf a, v16bf b, v8f c) {
    return __builtin_amdgcn_wmma_f32_16x16x32_bf16(false, a, false, b, (short)0, c, false, false);
}

// ---------------- kernel 1: f32 -> bf16 streaming convert --------------------
__global__ void cvt_f32_bf16(const float* __restrict__ src, unsigned short* __restrict__ dst,
                             size_t n4) {
    size_t stride = (size_t)gridDim.x * blockDim.x;
    for (size_t i = (size_t)blockIdx.x * blockDim.x + threadIdx.x; i < n4; i += stride) {
        float4 f = ((const float4*)src)[i];
        unsigned lo = (unsigned)f32_to_bf16(f.x) | ((unsigned)f32_to_bf16(f.y) << 16);
        unsigned hi = (unsigned)f32_to_bf16(f.z) | ((unsigned)f32_to_bf16(f.w) << 16);
        ((uint2*)dst)[i] = make_uint2(lo, hi);
    }
}

// ---------------- kernel 2: router (softmax top-2 + renorm scales) -----------
__global__ void moe_router(const float* __restrict__ logits,
                           int* __restrict__ tke, float* __restrict__ tks) {
    int t = blockIdx.x * blockDim.x + threadIdx.x;
    if (t >= T_) return;
    const float* l = logits + (size_t)t * E_;
    float m1 = -3.4e38f, m2 = -3.4e38f; int i1 = 0, i2 = 0;
    for (int j = 0; j < E_; ++j) {
        float v = l[j];
        if (v > m1)      { m2 = m1; i2 = i1; m1 = v; i1 = j; }
        else if (v > m2) { m2 = v;  i2 = j; }
    }
    float p2 = __expf(m2 - m1);
    float inv = 1.0f / (1.0f + p2);
    tke[t * 2] = i1;  tke[t * 2 + 1] = i2;
    tks[t * 2] = inv; tks[t * 2 + 1] = p2 * inv;
}

// ---------------- kernel 3: parallel deterministic capacity dispatch ---------
// One block per expert. Each thread owns a CONTIGUOUS 128-assignment range (so
// global order == flattened (token,k) order == jnp stable argsort order),
// counts matches, exclusive-scans the 256 counts in LDS, then re-scans writing
// slots at its scanned offset. Capacity drop: pos >= C skipped.
__global__ __launch_bounds__(256) void moe_dispatch(
        const int* __restrict__ tke, const float* __restrict__ tks,
        int* __restrict__ slot_tok, float* __restrict__ slot_sc,
        int* __restrict__ ecount) {
    const int e = blockIdx.x, t = threadIdx.x;
    constexpr int PER = (2 * T_) / 256;   // 128
    const int base = t * PER;
    __shared__ int cnts[256];
    __shared__ int total;

    int c = 0;
    for (int i = 0; i < PER; ++i) c += (tke[base + i] == e);
    cnts[t] = c;
    __syncthreads();
    if (t == 0) {                          // tiny serial exclusive scan (256)
        int run = 0;
        for (int j = 0; j < 256; ++j) { int x = cnts[j]; cnts[j] = run; run += x; }
        total = run;
    }
    __syncthreads();
    int pos = cnts[t];
    for (int i = 0; i < PER; ++i) {
        int a = base + i;
        if (tke[a] == e) {
            if (pos < C_) { slot_tok[e * C_ + pos] = a >> 1; slot_sc[e * C_ + pos] = tks[a]; }
            ++pos;
        }
    }
    if (t == 0) ecount[e] = total < C_ ? total : C_;
}

// ---------------- kernel 4: GEMM1 (gathered) + fused SwiGLU ------------------
// h[c,j] = sum_h buf[c,h]*w31[e,j,h] (NT, K=H). Block: M=128 slots x 64 SwiGLU
// cols; B staging carries matching h3 and h1 rows so each wave holds acc3/acc1
// for the SAME (row,col) and fuses act = h3*silu(h1) in registers.
// Double-buffered async pipeline, ONE barrier per K tile:
//   wait_asynccnt 0 ; barrier ; issue tile k+1 -> other buf ; 8x WMMA on tile k
__global__ __launch_bounds__(256) void moe_gemm1(
        const unsigned short* __restrict__ hid,   // [T,H] bf16
        const unsigned short* __restrict__ w31,   // [E,2I,H] bf16
        const int* __restrict__ slot_tok, const int* __restrict__ ecount,
        unsigned short* __restrict__ act)         // [E,C,I] bf16
{
    const int e = blockIdx.z, mt = blockIdx.y, nt = blockIdx.x;
    const int count = ecount[e];
    const int m0 = mt * 128;
    if (m0 >= count) return;

    constexpr int ATILE = 128 * 48, BTILE = 128 * 48;
    __shared__ alignas(16) unsigned short As[2 * ATILE];
    __shared__ alignas(16) unsigned short Bs[2 * BTILE];

    const int tid = threadIdx.x;
    const int co  = (tid & 3) * 8;            // 16B chunk within 32-elem K tile

    const unsigned short* asrc[2];
    const unsigned short* bsrc[2];
    int arow[2];
    for (int l = 0; l < 2; ++l) {
        int r = (tid >> 2) + l * 64;          // 0..127
        arow[l] = r;
        int g = m0 + r;
        asrc[l] = (g < count) ? hid + (size_t)slot_tok[e * C_ + g] * H_ : nullptr;
        int gj = (r < 64) ? (nt * 64 + r) : (I_ + nt * 64 + (r - 64));
        bsrc[l] = w31 + ((size_t)e * (2 * I_) + gj) * (size_t)H_;
    }
    // pre-zero over-capacity A rows in BOTH buffers (async issues skip them)
    for (int l = 0; l < 2; ++l)
        if (!asrc[l]) {
            uint4 z = make_uint4(0u, 0u, 0u, 0u);
            *(uint4*)(As + arow[l] * 48 + co) = z;
            *(uint4*)(As + ATILE + arow[l] * 48 + co) = z;
        }

    auto issue = [&](int kt, int buf) {
        int k0 = kt * 32;
        for (int l = 0; l < 2; ++l) {
            if (asrc[l]) stage16(asrc[l] + k0 + co, As + buf * ATILE + arow[l] * 48 + co);
            stage16(bsrc[l] + k0 + co, Bs + buf * BTILE + arow[l] * 48 + co);
        }
    };

    const int lane  = tid & 31;
    const int wv    = tid >> 5;               // 8 waves: 4(M) x 2(N)
    const int m_w   = (wv >> 1) * 32;
    const int n_w   = (wv & 1) * 32;
    const int lrow  = lane & 15;
    const int khalf = lane >> 4;

    v8f acc3[2][2] = {};
    v8f acc1[2][2] = {};

    constexpr int KT = H_ / 32;
    issue(0, 0);
    for (int kt = 0; kt < KT; ++kt) {
        const int cur = kt & 1;
        wait_async0();                         // own tile-kt loads landed
        __syncthreads();                       // everyone's landed; prev reads done
        if (kt + 1 < KT) issue(kt + 1, cur ^ 1);

        const unsigned short* Ab = As + cur * ATILE;
        const unsigned short* Bb = Bs + cur * BTILE;
        v16bf af[2], bf3[2], bf1[2];
        for (int sm = 0; sm < 2; ++sm) af[sm] = load_a_frag(Ab, m_w + sm * 16 + lrow, khalf);
        for (int sn = 0; sn < 2; ++sn) {
            bf3[sn] = load_b_frag(Bb, n_w + sn * 16 + lrow, khalf);
            bf1[sn] = load_b_frag(Bb, 64 + n_w + sn * 16 + lrow, khalf);
        }
        for (int sm = 0; sm < 2; ++sm)
            for (int sn = 0; sn < 2; ++sn) {
                acc3[sm][sn] = wmma_bf16(af[sm], bf3[sn], acc3[sm][sn]);
                acc1[sm][sn] = wmma_bf16(af[sm], bf1[sn], acc1[sm][sn]);
            }
    }

    // fused SwiGLU epilogue; C/D layout: element v -> M = v + 8*khalf, N = lane&15
    for (int sm = 0; sm < 2; ++sm)
        for (int sn = 0; sn < 2; ++sn)
            for (int v = 0; v < 8; ++v) {
                int row = m_w + sm * 16 + khalf * 8 + v;
                int col = nt * 64 + n_w + sn * 16 + lrow;
                float h3 = acc3[sm][sn][v];
                float h1 = acc1[sm][sn][v];
                float a  = h3 * (h1 / (1.0f + __expf(-h1)));
                act[(size_t)(e * C_ + m0 + row) * I_ + col] = f32_to_bf16(a);
            }
}

// ---------------- kernel 5: GEMM2 + scaled scatter-add -----------------------
__global__ __launch_bounds__(256) void moe_gemm2(
        const unsigned short* __restrict__ act,   // [E,C,I] bf16
        const unsigned short* __restrict__ w2b,   // [E,H,I] bf16
        const int* __restrict__ slot_tok, const float* __restrict__ slot_sc,
        const int* __restrict__ ecount, float* __restrict__ out)
{
    const int e = blockIdx.z, mt = blockIdx.y, ht = blockIdx.x;
    const int count = ecount[e];
    const int m0 = mt * 128;
    if (m0 >= count) return;

    constexpr int ATILE = 128 * 48, BTILE = 64 * 48;
    __shared__ alignas(16) unsigned short As[2 * ATILE];
    __shared__ alignas(16) unsigned short Bs[2 * BTILE];

    const int tid = threadIdx.x;
    const int co  = (tid & 3) * 8;

    // 768 16B chunks per K tile (A:512, B:256) -> 3 per thread
    const unsigned short* src[3];
    unsigned short* dstb[3];
    int dstride[3];
    for (int l = 0; l < 3; ++l) {
        int i = tid + l * 256;
        if (i < 512) {
            int r = i >> 2;
            src[l] = act + (size_t)(e * C_ + m0 + r) * I_;
            dstb[l] = As + r * 48 + co;  dstride[l] = ATILE;
        } else {
            int j = (i - 512) >> 2;
            src[l] = w2b + ((size_t)e * H_ + ht * 64 + j) * (size_t)I_;
            dstb[l] = Bs + j * 48 + co;  dstride[l] = BTILE;
        }
    }
    auto issue = [&](int kt, int buf) {
        int k0 = kt * 32;
        for (int l = 0; l < 3; ++l)
            stage16(src[l] + k0 + co, dstb[l] + buf * dstride[l]);
    };

    const int lane  = tid & 31;
    const int wv    = tid >> 5;
    const int m_w   = (wv >> 1) * 32;
    const int n_w   = (wv & 1) * 32;
    const int lrow  = lane & 15;
    const int khalf = lane >> 4;

    v8f acc[2][2] = {};

    constexpr int KT = I_ / 32;
    issue(0, 0);
    for (int kt = 0; kt < KT; ++kt) {
        const int cur = kt & 1;
        wait_async0();
        __syncthreads();
        if (kt + 1 < KT) issue(kt + 1, cur ^ 1);

        const unsigned short* Ab = As + cur * ATILE;
        const unsigned short* Bb = Bs + cur * BTILE;
        v16bf af[2], bf[2];
        for (int sm = 0; sm < 2; ++sm) af[sm] = load_a_frag(Ab, m_w + sm * 16 + lrow, khalf);
        for (int sn = 0; sn < 2; ++sn) bf[sn] = load_b_frag(Bb, n_w + sn * 16 + lrow, khalf);
        for (int sm = 0; sm < 2; ++sm)
            for (int sn = 0; sn < 2; ++sn)
                acc[sm][sn] = wmma_bf16(af[sm], bf[sn], acc[sm][sn]);
    }

    for (int sm = 0; sm < 2; ++sm)
        for (int sn = 0; sn < 2; ++sn)
            for (int v = 0; v < 8; ++v) {
                int row = m_w + sm * 16 + khalf * 8 + v;
                int g = m0 + row;
                if (g < count) {
                    int col   = ht * 64 + n_w + sn * 16 + lrow;
                    int tok   = slot_tok[e * C_ + g];
                    float scl = slot_sc[e * C_ + g];
                    atomicAdd(out + (size_t)tok * H_ + col, acc[sm][sn][v] * scl);
                }
            }
}

// ---------------- host: workspace carve + launch sequence --------------------
static inline size_t align256(size_t x) { return (x + 255) & ~(size_t)255; }

extern "C" void kernel_launch(void* const* d_in, const int* in_sizes, int n_in,
                              void* d_out, int out_size, void* d_ws, size_t ws_size,
                              hipStream_t stream) {
    (void)in_sizes; (void)n_in; (void)out_size; (void)ws_size;
    const float* hidden = (const float*)d_in[0];   // [T,H]
    const float* logits = (const float*)d_in[1];   // [T,E]
    const float* w31f   = (const float*)d_in[2];   // [E,2I,H]
    const float* w2f    = (const float*)d_in[3];   // [E,H,I]
    float* out = (float*)d_out;                    // [T,H]

    char* p = (char*)d_ws;
    size_t off = 0;
    int*   topk_e  = (int*)(p + off);            off = align256(off + sizeof(int)   * 2 * T_);
    float* topk_s  = (float*)(p + off);          off = align256(off + sizeof(float) * 2 * T_);
    int*   slot_t  = (int*)(p + off);            off = align256(off + sizeof(int)   * E_ * C_);
    float* slot_s  = (float*)(p + off);          off = align256(off + sizeof(float) * E_ * C_);
    int*   ecount  = (int*)(p + off);            off = align256(off + sizeof(int)   * E_);
    unsigned short* hid_bf = (unsigned short*)(p + off); off = align256(off + 2ull * T_ * H_);
    unsigned short* w31_bf = (unsigned short*)(p + off); off = align256(off + 2ull * E_ * 2 * I_ * H_);
    unsigned short* w2_bf  = (unsigned short*)(p + off); off = align256(off + 2ull * E_ * H_ * I_);
    unsigned short* act_bf = (unsigned short*)(p + off); off = align256(off + 2ull * E_ * C_ * I_);

    (void)hipMemsetAsync(d_out, 0, sizeof(float) * (size_t)T_ * H_, stream);

    cvt_f32_bf16<<<2048, 256, 0, stream>>>(hidden, hid_bf, (size_t)T_ * H_ / 4);
    cvt_f32_bf16<<<8192, 256, 0, stream>>>(w31f,  w31_bf, (size_t)E_ * 2 * I_ * H_ / 4);
    cvt_f32_bf16<<<4096, 256, 0, stream>>>(w2f,   w2_bf,  (size_t)E_ * H_ * I_ / 4);

    moe_router<<<T_ / 256, 256, 0, stream>>>(logits, topk_e, topk_s);
    moe_dispatch<<<E_, 256, 0, stream>>>(topk_e, topk_s, slot_t, slot_s, ecount);

    moe_gemm1<<<dim3(I_ / 64, C_ / 128, E_), 256, 0, stream>>>(hid_bf, w31_bf, slot_t, ecount, act_bf);
    moe_gemm2<<<dim3(H_ / 64, C_ / 128, E_), 256, 0, stream>>>(act_bf, w2_bf, slot_t, slot_s, ecount, out);
}
